// ConvModLayer_27367531610765
// MI455X (gfx1250) — matrix operand order repaired and checked by reading, not compiled, not run on hardware
//
#include <hip/hip_runtime.h>
#include <math.h>

typedef __attribute__((ext_vector_type(16))) __bf16 v16bf;
typedef __attribute__((ext_vector_type(8)))  float  v8f;

// Problem sizes
constexpr int BB = 16, CC = 512, HH = 64, WW = 64;
constexpr float EPS = 1e-8f;

// Workspace layout (bytes)
constexpr size_t XS_OFF  = 0;                                  // bf16 [16][512][64][64]
constexpr size_t WT_OFF  = (size_t)BB * CC * HH * WW * 2;      // bf16 [9][512][512]
constexpr size_t WSQ_OFF = WT_OFF + (size_t)9 * CC * CC * 2;   // f32  [512][512]
constexpr size_t SIG_OFF = WSQ_OFF + (size_t)CC * CC * 4;      // f32  [16][512]

// ---------------------------------------------------------------------------
// Prep 1: xs[b,i,h,w] = bf16( x[b,i,h,w] * s[b,i] )
// ---------------------------------------------------------------------------
__global__ void k_scale_x(const float* __restrict__ x, const float* __restrict__ s,
                          __bf16* __restrict__ xs) {
    size_t t    = (size_t)blockIdx.x * blockDim.x + threadIdx.x; // handles 4 elems
    size_t base = t * 4;
    int ch = (int)(base >> 12);            // element / 4096
    int b  = ch >> 9, i = ch & 511;
    float sv = s[b * 512 + i];
    float4 v = *(const float4*)(x + base);
    union { __bf16 h[4]; uint2 u; } o;
    o.h[0] = (__bf16)(v.x * sv);
    o.h[1] = (__bf16)(v.y * sv);
    o.h[2] = (__bf16)(v.z * sv);
    o.h[3] = (__bf16)(v.w * sv);
    *(uint2*)(xs + base) = o.u;
}

// ---------------------------------------------------------------------------
// Prep 2: wT[kp][o][i] = bf16( c * w[o][i][kp] ),  c = 1/sqrt(512*9)
// ---------------------------------------------------------------------------
__global__ void k_wt(const float* __restrict__ w, __bf16* __restrict__ wT) {
    int idx = blockIdx.x * 256 + threadIdx.x;        // 9*512*512
    int kp  = idx / (512 * 512);
    int rem = idx & (512 * 512 - 1);
    int o   = rem >> 9, i = rem & 511;
    const float c = 1.0f / sqrtf(4608.0f);
    wT[(size_t)kp * 512 * 512 + rem] = (__bf16)(w[((size_t)o * 512 + i) * 9 + kp] * c);
}

// ---------------------------------------------------------------------------
// Prep 3: wsq[o][i] = sum_k w[o][i][k]^2
// ---------------------------------------------------------------------------
__global__ void k_wsq(const float* __restrict__ w, float* __restrict__ wsq) {
    int idx = blockIdx.x * 256 + threadIdx.x;        // 512*512
    const float* p = w + (size_t)idx * 9;
    float a = 0.f;
#pragma unroll
    for (int j = 0; j < 9; j++) a += p[j] * p[j];
    wsq[idx] = a;
}

// ---------------------------------------------------------------------------
// Prep 4: sig[b][o] = rsqrt( c^2 * sum_i s[b,i]^2 * wsq[o,i] + eps )
// ---------------------------------------------------------------------------
__global__ void k_sigma(const float* __restrict__ s, const float* __restrict__ wsq,
                        float* __restrict__ sig) {
    int idx = blockIdx.x * 256 + threadIdx.x;        // 16*512
    int b = idx >> 9, o = idx & 511;
    const float* sp = s + b * 512;
    const float* wp = wsq + (size_t)o * 512;
    float a = 0.f;
    for (int i = 0; i < 512; i++) { float sv = sp[i]; a += sv * sv * wp[i]; }
    sig[idx] = rsqrtf(a * (1.0f / 4608.0f) + EPS);
}

// ---------------------------------------------------------------------------
// Main: implicit-GEMM 3x3 conv via v_wmma_f32_16x16x32_bf16, double-buffered.
// grid = (32 row-pairs, 4 o-tiles, 16 batches), block = 256 (8 waves, 4M x 2N)
// Block tile: 128 (o) x 128 (two image rows), K chunks of 32 channels per tap,
// 144 flattened K-iterations, LDS ping-pong, one barrier per iteration.
// ---------------------------------------------------------------------------
struct Stage { uint4 a0, a1, b0, b1; };

__global__ __launch_bounds__(256) void k_conv(const __bf16* __restrict__ xs,
                                              const __bf16* __restrict__ wT,
                                              const float* __restrict__ sig,
                                              float* __restrict__ out) {
    __shared__ __align__(16) __bf16 As[2][128][40];    // 80B rows, ping-pong
    __shared__ __align__(16) __bf16 Bs[2][2][66][40];  // [buf][rowgroup][n+1 halo][k]

    const int y0 = blockIdx.x * 2;
    const int o0 = blockIdx.y * 128;
    const int b  = blockIdx.z;
    const int t  = threadIdx.x;
    const int lane = t & 31;
    const int wid  = t >> 5;
    const int wm   = wid & 3;      // wave M position (0..3)
    const int wn   = wid >> 2;     // wave N position (0..1)
    const int lr   = lane & 15;
    const int hi   = lane >> 4;

    // staging thread roles
    const int arow = t >> 1, apart = t & 1;      // A tile: 32B per thread
    const int kk = t >> 3;                       // B tile: channel 0..31
    const int xg = (t & 7) << 3;                 // B tile: pixel group base

    const __bf16* xsb = xs + (size_t)b * 512 * 4096;

    v8f acc[2][2][2];              // [mi][rowgroup][ni]
#pragma unroll
    for (int mi = 0; mi < 2; mi++)
#pragma unroll
        for (int rg = 0; rg < 2; rg++)
#pragma unroll
            for (int ni = 0; ni < 2; ni++) { v8f z = {}; acc[mi][rg][ni] = z; }

    // ---- global -> regs for iteration `it` (branch-free border handling) ----
    auto stage_load = [&](int it) -> Stage {
        Stage st;
        int kp = it >> 4, ic = it & 15;
        int dy = kp / 3 - 1;
        const __bf16* gp = wT + ((size_t)kp << 18) + (size_t)(o0 + arow) * 512
                              + (ic << 5) + apart * 16;
        st.a0 = ((const uint4*)gp)[0];
        st.a1 = ((const uint4*)gp)[1];
        __builtin_prefetch(gp + 64, 0, 1);       // A stream, two chunks ahead
#pragma unroll
        for (int rg = 0; rg < 2; rg++) {
            int ys  = y0 + rg + dy;
            int ysc = ys < 0 ? 0 : (ys > 63 ? 63 : ys);   // always-valid address
            uint4 v = *(const uint4*)(xsb + ((size_t)((ic << 5) + kk) << 12)
                                          + (ysc << 6) + xg);
            if ((unsigned)ys >= 64u) { v.x = 0; v.y = 0; v.z = 0; v.w = 0; }
            if (rg) st.b1 = v; else st.b0 = v;
        }
        return st;
    };

    // ---- regs -> LDS buffer q for iteration `it` ----
    auto stage_store = [&](int q, int it, const Stage& st) {
        int kp = it >> 4;
        int dx = kp - (kp / 3) * 3 - 1;
        uint4* d = (uint4*)((char*)&As[q][arow][0] + apart * 32);
        d[0] = st.a0; d[1] = st.a1;
#pragma unroll
        for (int rg = 0; rg < 2; rg++) {
            union { uint4 u; __bf16 h[8]; } v;
            v.u = rg ? st.b1 : st.b0;
            __bf16* bp = &Bs[q][rg][xg + 1 - dx][kk];
#pragma unroll
            for (int j = 0; j < 8; j++) bp[j * 40] = v.h[j];
            if (dx != 0 && t < 32)               // edge column never scattered
                Bs[q][rg][dx == 1 ? 64 : 1][t] = (__bf16)0.0f;
        }
    };

    // ---- one pipelined iteration: compute from buf q, stage it+1 into q^1 ----
    auto body = [&](int it, int q) {
        int itn = (it + 1 < 144) ? it + 1 : 143;
        Stage nx = stage_load(itn);              // long-latency loads first

        union U16 { uint4 u[2]; v16bf v; };
        U16 af[2], bf[2][2];
#pragma unroll
        for (int mi = 0; mi < 2; mi++) {
            // A 16x32 bf16: lanes 0-15 K{0..7,16..23}, lanes 16-31 K{8..15,24..31}
            const char* ap = (const char*)&As[q][wm * 32 + mi * 16 + lr][0];
            int c0 = hi ? 16 : 0;
            af[mi].u[0] = *(const uint4*)(ap + c0);
            af[mi].u[1] = *(const uint4*)(ap + c0 + 32);
        }
#pragma unroll
        for (int rg = 0; rg < 2; rg++)
#pragma unroll
            for (int ni = 0; ni < 2; ni++) {
                // B 32x16 bf16: lanes 0-15 K{0..7,16..23}, lanes 16-31 K{8..15,24..31}
                const char* bp = (const char*)&Bs[q][rg][1 + wn * 32 + ni * 16 + lr][0];
                int c0 = hi ? 16 : 0;
                bf[rg][ni].u[0] = *(const uint4*)(bp + c0);
                bf[rg][ni].u[1] = *(const uint4*)(bp + c0 + 32);
            }

#pragma unroll
        for (int mi = 0; mi < 2; mi++)
#pragma unroll
            for (int rg = 0; rg < 2; rg++)
#pragma unroll
                for (int ni = 0; ni < 2; ni++)
                    acc[mi][rg][ni] = __builtin_amdgcn_wmma_f32_16x16x32_bf16(
                        false, af[mi].v, false, bf[rg][ni].v,
                        (short)0, acc[mi][rg][ni], false, false);

        stage_store(q ^ 1, itn, nx);             // overlaps with WMMA drain
        __syncthreads();
    };

    // ---- prologue + pipelined main loop (buffer index statically known) ----
    {
        Stage st = stage_load(0);
        stage_store(0, 0, st);
        __syncthreads();
    }
#pragma unroll 1
    for (int it = 0; it < 144; it += 2) {
        body(it,     0);
        body(it + 1, 1);
    }

    // ---- epilogue: demodulate and store f32 ----
#pragma unroll
    for (int mi = 0; mi < 2; mi++) {
#pragma unroll
        for (int r = 0; r < 8; r++) {
            int o = o0 + wm * 32 + mi * 16 + r + (hi ? 8 : 0);
            float sg = sig[b * 512 + o];
            float* op = out + (((size_t)b * 512 + o) << 12) + (y0 << 6);
#pragma unroll
            for (int rg = 0; rg < 2; rg++)
#pragma unroll
                for (int ni = 0; ni < 2; ni++)
                    op[(rg << 6) + wn * 32 + ni * 16 + lr] = acc[mi][rg][ni][r] * sg;
        }
    }
}

// ---------------------------------------------------------------------------
extern "C" void kernel_launch(void* const* d_in, const int* in_sizes, int n_in,
                              void* d_out, int out_size, void* d_ws, size_t ws_size,
                              hipStream_t stream) {
    const float* x = (const float*)d_in[0];   // [16,512,64,64]
    const float* s = (const float*)d_in[1];   // [16,512]
    const float* w = (const float*)d_in[2];   // [512,512,3,3]
    float* out = (float*)d_out;

    char* ws = (char*)d_ws;
    __bf16* xs  = (__bf16*)(ws + XS_OFF);
    __bf16* wT  = (__bf16*)(ws + WT_OFF);
    float*  wsq = (float*)(ws + WSQ_OFF);
    float*  sig = (float*)(ws + SIG_OFF);

    k_scale_x<<<(BB * CC * HH * WW / 4) / 256, 256, 0, stream>>>(x, s, xs);
    k_wt     <<<(9 * CC * CC) / 256,           256, 0, stream>>>(w, wT);
    k_wsq    <<<(CC * CC) / 256,               256, 0, stream>>>(w, wsq);
    k_sigma  <<<(BB * CC) / 256,               256, 0, stream>>>(s, wsq, sig);
    k_conv   <<<dim3(HH / 2, CC / 128, BB),    256, 0, stream>>>(xs, wT, sig, out);
}
